// SingleLayerTransformer_60198261620997
// MI455X (gfx1250) — matrix operand order repaired and checked by reading, not compiled, not run on hardware
//
#include <hip/hip_runtime.h>

// CDNA5 / gfx1250 single-layer transformer forward.
// Pass 1: prep weights -> f16, pre-swizzled into WMMA B-fragment layout in d_ws.
// Pass 2: per-wave 16-row batch tiles, all contractions via v_wmma_f32_16x16x32_f16.
// 8 waves / 256 threads per block, 303 KB LDS -> 2 waves per SIMD32 for latency hiding.
typedef __attribute__((ext_vector_type(16))) _Float16 v16h;
typedef __attribute__((ext_vector_type(8)))  _Float16 v8h;
typedef __attribute__((ext_vector_type(8)))  float    v8f;

#define DMODEL 128
#define DMLP   512
#define NVOCAB 114
#define WIN    3
#define TILE_M 16
#define WAVES  8

// ---- workspace fragment segments (units: f16 halves; 512 halves = 1 fragment) ----
// fragment = 32 lanes x 16 halves, lane-major: ws[frag*512 + lane*16 + e]
#define WK_OFF   0        // 32 frags (nt*4+kb)
#define WQ_OFF   16384    // 32
#define WV_OFF   32768    // 32
#define WO_OFF   49152    // 32
#define WIN_OFF  65536    // 128 (nt*4+kb), nt<32
#define WOUT_OFF 131072   // 128 (nt*16+kb), kb<16
#define UNE_OFF  196608   // 32 (masked cols >=114)
#define N_FRAGS  416

// ---- per-wave LDS slice layout (bytes) ----
#define OFF_X0H   0            // [3][16][128] f16   12288  (dead after stage 4)
#define OFF_X1F   0            // [16][128]    f32    8192  (alias over x0h[t=0..1])
#define OFF_KBUF  12288        // [3][16][128] f16   12288  (aliased by MLPH later)
#define OFF_MLPH  12288        // [16][512]    f16   16384  (alias of KBUF+QBUF)
#define OFF_QBUF  24576        // [16][128]    f16    4096
#define OFF_SCORE 28672        // [16][4][3]   f32    1024
#define OFF_VH    29696        // [16][128]    f16    4096  (reused as x2h)
#define OFF_X1H   33792        // [16][128]    f16    4096
#define PER_WAVE  37888        // x8 waves = 303104 B < 320 KB

__device__ __forceinline__ v8f zero8() {
  v8f z;
#pragma unroll
  for (int i = 0; i < 8; ++i) z[i] = 0.0f;
  return z;
}

__device__ __forceinline__ v8f wmma16(v16h a, v16h b, v8f c) {
  return __builtin_amdgcn_wmma_f32_16x16x32_f16(false, a, false, b, (short)0, c,
                                                false, false);
}

// A fragment (16x32 f16) from row-major LDS tile [16][ld]; two 16B vector loads.
__device__ __forceinline__ v16h load_A_lds(const _Float16* T, int ld, int k0, int lane) {
  const int m  = lane & 15;
  const int hb = (lane >> 4) << 4;
  const _Float16* row = T + m * ld + k0 + hb;
  v8h lo = *(const v8h*)(row);
  v8h hi = *(const v8h*)(row + 8);
  return __builtin_shufflevector(lo, hi, 0, 1, 2, 3, 4, 5, 6, 7,
                                 8, 9, 10, 11, 12, 13, 14, 15);
}

// B fragment: one contiguous 32B load from pre-swizzled workspace.
__device__ __forceinline__ v16h load_B_ws(const _Float16* __restrict__ ws,
                                          int fragHalfOff, int lane) {
  return *(const v16h*)(ws + fragHalfOff + lane * 16);
}

// D/C layout: VGPR r holds row (r + 8*(lane>>4)), column (lane&15)+nt*16.
__device__ __forceinline__ void store_D_h16(_Float16* T, int ld, int nt, v8f d, int lane) {
  const int col = nt * 16 + (lane & 15);
  const int rb  = (lane >> 4) << 3;
#pragma unroll
  for (int r = 0; r < 8; ++r) T[(rb + r) * ld + col] = (_Float16)d[r];
}

__device__ __forceinline__ void store_D_f32(float* T, int ld, int nt, v8f d, int lane) {
  const int col = nt * 16 + (lane & 15);
  const int rb  = (lane >> 4) << 3;
#pragma unroll
  for (int r = 0; r < 8; ++r) T[(rb + r) * ld + col] = d[r];
}

__device__ __forceinline__ v8f load_C_f32(const float* T, int ld, int nt, int lane) {
  const int col = nt * 16 + (lane & 15);
  const int rb  = (lane >> 4) << 3;
  v8f c;
#pragma unroll
  for (int r = 0; r < 8; ++r) c[r] = T[(rb + r) * ld + col];
  return c;
}

// C init from an f16 tile (residual read back from LDS f16).
__device__ __forceinline__ v8f load_C_h16(const _Float16* T, int ld, int nt, int lane) {
  const int col = nt * 16 + (lane & 15);
  const int rb  = (lane >> 4) << 3;
  v8f c;
#pragma unroll
  for (int r = 0; r < 8; ++r) c[r] = (float)T[(rb + r) * ld + col];
  return c;
}

// K=128 chain with 2 accumulators (breaks WMMA D->C serial dependency).
__device__ __forceinline__ v8f gemm4(v16h a0, v16h a1, v16h a2, v16h a3,
                                     const _Float16* __restrict__ ws, int fragBase,
                                     int lane, v8f cinit) {
  v8f c0 = wmma16(a0, load_B_ws(ws, fragBase + 0 * 512, lane), cinit);
  v8f c1 = wmma16(a1, load_B_ws(ws, fragBase + 1 * 512, lane), zero8());
  c0 = wmma16(a2, load_B_ws(ws, fragBase + 2 * 512, lane), c0);
  c1 = wmma16(a3, load_B_ws(ws, fragBase + 3 * 512, lane), c1);
#pragma unroll
  for (int r = 0; r < 8; ++r) c0[r] += c1[r];
  return c0;
}

// ---------------- Pass 1: weight conversion + B-layout swizzle ----------------
__global__ void __launch_bounds__(32)
prep_weights(const float* __restrict__ Wk, const float* __restrict__ Wq,
             const float* __restrict__ Wv, const float* __restrict__ Wo,
             const float* __restrict__ Win, const float* __restrict__ Wout,
             const float* __restrict__ Un, _Float16* __restrict__ ws) {
  const int f = blockIdx.x;
  const int lane = threadIdx.x;
  const int nlo = lane & 15;
  const int khi = (lane >> 4) << 4;
  _Float16* dst = ws + (size_t)f * 512 + lane * 16;
  const float* src;
  bool zero = false;
  if (f < 32) {                       // W_k flat [128][128]
    const int nt = f >> 2, kb = f & 3;
    src = Wk + (nt * 16 + nlo) * DMODEL + kb * 32 + khi;
  } else if (f < 64) {                // W_q
    const int g = f - 32, nt = g >> 2, kb = g & 3;
    src = Wq + (nt * 16 + nlo) * DMODEL + kb * 32 + khi;
  } else if (f < 96) {                // W_v
    const int g = f - 64, nt = g >> 2, kb = g & 3;
    src = Wv + (nt * 16 + nlo) * DMODEL + kb * 32 + khi;
  } else if (f < 128) {               // W_o[h][m][d] as B[k=hd][n=m]
    const int g = f - 96, nt = g >> 2, kb = g & 3;
    const int n = nt * 16 + nlo, k0 = kb * 32 + khi;
    src = Wo + (k0 >> 5) * (DMODEL * 32) + n * 32 + (k0 & 31);
  } else if (f < 256) {               // W_in [512][128]
    const int g = f - 128, nt = g >> 2, kb = g & 3;
    src = Win + (nt * 16 + nlo) * DMODEL + kb * 32 + khi;
  } else if (f < 384) {               // W_out [128][512] as B[k=p][n=m]
    const int g = f - 256, nt = g >> 4, kb = g & 15;
    src = Wout + (nt * 16 + nlo) * DMLP + kb * 32 + khi;
  } else {                            // unembed [114][128], mask n>=114
    const int g = f - 384, nt = g >> 2, kb = g & 3;
    const int n = nt * 16 + nlo;
    zero = (n >= NVOCAB);
    src = Un + (zero ? (NVOCAB - 1) : n) * DMODEL + kb * 32 + khi;
  }
#pragma unroll
  for (int e = 0; e < 16; ++e)
    dst[e] = zero ? (_Float16)0.0f : (_Float16)src[e];
}

// ---------------- Pass 2: forward ----------------
__global__ void __launch_bounds__(WAVES * 32, 1)
slt_fwd_wmma(const int* __restrict__ xs, const float* __restrict__ embed,
             const float* __restrict__ pose, const _Float16* __restrict__ ws,
             float* __restrict__ out) {
  extern __shared__ __align__(16) char smem[];
  const int lane = threadIdx.x & 31;
  const int wave = threadIdx.x >> 5;
  const int r0   = (blockIdx.x * WAVES + wave) * TILE_M;

  char* base = smem + wave * PER_WAVE;
  _Float16* x0h  = (_Float16*)(base + OFF_X0H);
  _Float16* kbuf = (_Float16*)(base + OFF_KBUF);
  _Float16* qbuf = (_Float16*)(base + OFF_QBUF);
  float*    scrb = (float*)   (base + OFF_SCORE);
  _Float16* vh   = (_Float16*)(base + OFF_VH);     // later reused as x2h
  _Float16* x1h  = (_Float16*)(base + OFF_X1H);
  float*    x1f  = (float*)   (base + OFF_X1F);    // aliases dead x0h[t=0..1]
  _Float16* mlph = (_Float16*)(base + OFF_MLPH);   // aliases dead kbuf+qbuf

  const int nlo = lane & 15;
  const int rb  = (lane >> 4) << 3;

  // ---- Stage 1: x0 = embed[tok] + pos -> f16 LDS ----
  for (int i = lane; i < TILE_M * WIN * (DMODEL / 4); i += 32) {
    const int m4 = i & 31;
    const int rt = i >> 5;
    const int t = rt % WIN, r = rt / WIN;
    const int tok = xs[(r0 + r) * WIN + t];
    const float4 ev = *(const float4*)(embed + tok * DMODEL + m4 * 4);
    const float4 pv = *(const float4*)(pose + t * DMODEL + m4 * 4);
    _Float16* dst = x0h + (t * TILE_M + r) * DMODEL + m4 * 4;
    dst[0] = (_Float16)(ev.x + pv.x);
    dst[1] = (_Float16)(ev.y + pv.y);
    dst[2] = (_Float16)(ev.z + pv.z);
    dst[3] = (_Float16)(ev.w + pv.w);
  }

  // ---- Stage 2: k[t] = x0[t] @ Wk^T; q = x0[2] @ Wq^T (A fragments hoisted) ----
  for (int t = 0; t < WIN; ++t) {
    const _Float16* xt = x0h + t * TILE_M * DMODEL;
    const v16h a0 = load_A_lds(xt, DMODEL, 0, lane);
    const v16h a1 = load_A_lds(xt, DMODEL, 32, lane);
    const v16h a2 = load_A_lds(xt, DMODEL, 64, lane);
    const v16h a3 = load_A_lds(xt, DMODEL, 96, lane);
    for (int nt = 0; nt < 8; ++nt)
      store_D_h16(kbuf + t * TILE_M * DMODEL, DMODEL, nt,
                  gemm4(a0, a1, a2, a3, ws, WK_OFF + nt * 4 * 512, lane, zero8()),
                  lane);
    if (t == WIN - 1) {
      for (int nt = 0; nt < 8; ++nt)
        store_D_h16(qbuf, DMODEL, nt,
                    gemm4(a0, a1, a2, a3, ws, WQ_OFF + nt * 4 * 512, lane, zero8()),
                    lane);
    }
  }

  // ---- Stage 3: scores + softmax over t (per row,head) ----
  for (int p = lane; p < TILE_M * 4; p += 32) {
    const int r = p >> 2, h = p & 3;
    const _Float16* qrow = qbuf + r * DMODEL + h * 32;
    float s[WIN];
    for (int t = 0; t < WIN; ++t) {
      const _Float16* krow = kbuf + (t * TILE_M + r) * DMODEL + h * 32;
      float acc = 0.0f;
#pragma unroll
      for (int d = 0; d < 32; ++d) acc += (float)krow[d] * (float)qrow[d];
      s[t] = acc;
    }
    const float mx = fmaxf(s[0], fmaxf(s[1], s[2]));
    const float e0 = __expf(s[0] - mx), e1 = __expf(s[1] - mx), e2 = __expf(s[2] - mx);
    const float inv = 1.0f / (e0 + e1 + e2);
    float* sp = scrb + (r * 4 + h) * WIN;
    sp[0] = e0 * inv; sp[1] = e1 * inv; sp[2] = e2 * inv;
  }

  // ---- Stage 4: v_cat = sum_t A[t] * (x0[t] @ Wv^T); A scale applied on D frags ----
  {
    v16h xa[WIN][4];
#pragma unroll
    for (int t = 0; t < WIN; ++t) {
      const _Float16* xt = x0h + t * TILE_M * DMODEL;
#pragma unroll
      for (int kb = 0; kb < 4; ++kb) xa[t][kb] = load_A_lds(xt, DMODEL, kb * 32, lane);
    }
    for (int nt = 0; nt < 8; ++nt) {
      const int h = (nt * 16 + nlo) >> 5;
      v8f vacc = zero8();
#pragma unroll
      for (int t = 0; t < WIN; ++t) {
        v8f acc = gemm4(xa[t][0], xa[t][1], xa[t][2], xa[t][3], ws,
                        WV_OFF + nt * 4 * 512, lane, zero8());
#pragma unroll
        for (int r = 0; r < 8; ++r)
          vacc[r] += scrb[((rb + r) * 4 + h) * WIN + t] * acc[r];
      }
      store_D_h16(vh, DMODEL, nt, vacc, lane);
    }
  }

  // ---- Stage 5: x1 = v_cat @ Wo + x0_last ----
  // Residual C read from f16 x0h[t=2] (bytes 8192..12288); x1f alias occupies 0..8192,
  // so the f32 x1 stores do not clobber the residual source. x0h[t<2] is dead here.
  {
    const v16h a0 = load_A_lds(vh, DMODEL, 0, lane);
    const v16h a1 = load_A_lds(vh, DMODEL, 32, lane);
    const v16h a2 = load_A_lds(vh, DMODEL, 64, lane);
    const v16h a3 = load_A_lds(vh, DMODEL, 96, lane);
    const _Float16* x0last = x0h + (WIN - 1) * TILE_M * DMODEL;
    for (int nt = 0; nt < 8; ++nt) {
      v8f c = gemm4(a0, a1, a2, a3, ws, WO_OFF + nt * 4 * 512, lane,
                    load_C_h16(x0last, DMODEL, nt, lane));
      store_D_f32(x1f, DMODEL, nt, c, lane);
      store_D_h16(x1h, DMODEL, nt, c, lane);
    }
  }

  // ---- Stage 6: MLP = relu(x1 @ Win^T) [16x512] ----
  {
    const v16h a0 = load_A_lds(x1h, DMODEL, 0, lane);
    const v16h a1 = load_A_lds(x1h, DMODEL, 32, lane);
    const v16h a2 = load_A_lds(x1h, DMODEL, 64, lane);
    const v16h a3 = load_A_lds(x1h, DMODEL, 96, lane);
    for (int nt = 0; nt < DMLP / 16; ++nt) {
      v8f acc = gemm4(a0, a1, a2, a3, ws, WIN_OFF + nt * 4 * 512, lane, zero8());
#pragma unroll
      for (int r = 0; r < 8; ++r) acc[r] = fmaxf(acc[r], 0.0f);
      store_D_h16(mlph, DMLP, nt, acc, lane);
    }
  }

  // ---- Stage 7: x2 = MLP @ Wout^T + x1 (residual via C); all 16 A frags in regs ----
  {
    v16h m[16];
#pragma unroll
    for (int kb = 0; kb < 16; ++kb) m[kb] = load_A_lds(mlph, DMLP, kb * 32, lane);
    for (int nt = 0; nt < 8; ++nt) {
      v8f c0 = load_C_f32(x1f, DMODEL, nt, lane);
      v8f c1 = zero8();
#pragma unroll
      for (int kb = 0; kb < 16; kb += 2) {
        c0 = wmma16(m[kb],     load_B_ws(ws, WOUT_OFF + (nt * 16 + kb) * 512, lane), c0);
        c1 = wmma16(m[kb + 1], load_B_ws(ws, WOUT_OFF + (nt * 16 + kb + 1) * 512, lane), c1);
      }
#pragma unroll
      for (int r = 0; r < 8; ++r) c0[r] += c1[r];
      store_D_h16(vh, DMODEL, nt, c0, lane);   // vh reused as x2h
    }
  }

  // ---- Stage 8: logits = x2 @ unembed^T, store 114 valid columns ----
  {
    const v16h a0 = load_A_lds(vh, DMODEL, 0, lane);
    const v16h a1 = load_A_lds(vh, DMODEL, 32, lane);
    const v16h a2 = load_A_lds(vh, DMODEL, 64, lane);
    const v16h a3 = load_A_lds(vh, DMODEL, 96, lane);
    for (int nt = 0; nt < 8; ++nt) {
      v8f acc = gemm4(a0, a1, a2, a3, ws, UNE_OFF + nt * 4 * 512, lane, zero8());
      const int col = nt * 16 + nlo;
      if (col < NVOCAB) {
#pragma unroll
        for (int r = 0; r < 8; ++r)
          out[(size_t)(r0 + rb + r) * NVOCAB + col] = acc[r];
      }
    }
  }
}

extern "C" void kernel_launch(void* const* d_in, const int* in_sizes, int n_in,
                              void* d_out, int out_size, void* d_ws, size_t ws_size,
                              hipStream_t stream) {
  (void)n_in; (void)ws_size; (void)out_size;
  const int*   xs    = (const int*)d_in[0];
  const float* embed = (const float*)d_in[1];
  const float* pose  = (const float*)d_in[2];
  const float* Wk    = (const float*)d_in[3];
  const float* Wq    = (const float*)d_in[4];
  const float* Wv    = (const float*)d_in[5];
  const float* Wo    = (const float*)d_in[6];
  const float* Win   = (const float*)d_in[7];
  const float* Wout  = (const float*)d_in[8];
  const float* Unemb = (const float*)d_in[9];
  float*     out = (float*)d_out;
  _Float16*  wsf = (_Float16*)d_ws;   // needs 416 KB scratch

  prep_weights<<<dim3(N_FRAGS), dim3(32), 0, stream>>>(Wk, Wq, Wv, Wo, Win, Wout,
                                                       Unemb, wsf);

  const int batch  = in_sizes[0] / WIN;            // 131072
  const int blocks = batch / (TILE_M * WAVES);     // 1024
  const size_t shmem = (size_t)WAVES * PER_WAVE;   // 303104 B (< 320 KB WGP LDS)
  slt_fwd_wmma<<<dim3(blocks), dim3(WAVES * 32), shmem, stream>>>(xs, embed, pose,
                                                                  wsf, out);
}
